// RNNTStep_446676599235
// MI455X (gfx1250) — matrix operand order repaired and checked by reading, not compiled, not run on hardware
//
#include <hip/hip_runtime.h>
#include <hip/hip_bf16.h>

// ---------------------------------------------------------------------------
// RNN-T decode step for MI455X (gfx1250, wave32, WMMA).
// bf16 WMMA (v_wmma_f32_16x16x32_bf16, fp32 accum) for all GEMMs; fp32 for
// gate nonlinearities / cell state. Global->LDS staging uses CDNA5 async
// copies (global_load_async_to_lds_b128 + s_wait_asynccnt) with LDS
// double-buffering so the DMA overlaps the matrix pipe.
// ---------------------------------------------------------------------------

typedef __bf16 bf16_t;
typedef __attribute__((ext_vector_type(16))) __bf16 v16bf;
typedef __attribute__((ext_vector_type(8)))  __bf16 v8bf;
typedef __attribute__((ext_vector_type(8)))  float  v8f;

#define BM 128   // C-tile rows per block
#define BN 128   // C-tile cols per block
#define BK 64    // K per LDS stage (2 WMMA k-steps)
#define BKP 72   // padded LDS row stride (bank-conflict relief, keeps 16B align)

__device__ __forceinline__ v16bf frag_cat(v8bf lo, v8bf hi) {
  return __builtin_shufflevector(lo, hi, 0,1,2,3,4,5,6,7,8,9,10,11,12,13,14,15);
}

// Async 16B global -> LDS copy, tracked by ASYNCcnt.
__device__ __forceinline__ void async_copy_b128(unsigned lds_off, const void* gptr) {
  asm volatile("global_load_async_to_lds_b128 %0, %1, off"
               :: "v"(lds_off), "v"(gptr) : "memory");
}
__device__ __forceinline__ void wait_async0() {
  asm volatile("s_wait_asynccnt 0x0" ::: "memory");
}

// C[M,N] = act( A1[M,K1]*W1[N,K1]^T + A2[M,K2]*W2[N,K2]^T + b1 + b2 )
// A*, W* are bf16 row-major; C written fp32 (Cf) and/or bf16 (Cb).
__global__ __launch_bounds__(256)
void gemm_dual_kernel(const bf16_t* __restrict__ A1, const bf16_t* __restrict__ W1, int K1,
                      const bf16_t* __restrict__ A2, const bf16_t* __restrict__ W2, int K2,
                      const float* __restrict__ b1, const float* __restrict__ b2,
                      float* __restrict__ Cf, bf16_t* __restrict__ Cb,
                      int M, int N, int relu_act)
{
  __shared__ __align__(16) bf16_t Alds[2][BM * BKP];   // 2 x 18 KB
  __shared__ __align__(16) bf16_t Wlds[2][BN * BKP];   // 2 x 18 KB

  const int tid  = threadIdx.x;
  const int lane = tid & 31;
  const int wave = tid >> 5;       // 8 waves
  const int wm   = wave & 3;       // 4 waves down M (32 rows each)
  const int wn   = wave >> 2;      // 2 waves across N (64 cols each)
  const int bm   = blockIdx.y * BM;
  const int bn   = blockIdx.x * BN;
  const int hl   = lane >> 4;      // half-wave select (ISA fragment layout)
  const int l15  = lane & 15;

  const v8f vzero = {0.f,0.f,0.f,0.f,0.f,0.f,0.f,0.f};
  v8f acc[2][4];
  #pragma unroll
  for (int i = 0; i < 2; ++i)
    #pragma unroll
    for (int j = 0; j < 4; ++j) acc[i][j] = vzero;

  const int nst1 = K1 / BK;
  const int nst2 = A2 ? (K2 / BK) : 0;
  const int nst  = nst1 + nst2;

  // issue async copies for stage s into LDS buffer `buf`
  auto issue_stage = [&](int s, int buf) {
    const bf16_t* A; const bf16_t* W; int K; int kt;
    if (s < nst1) { A = A1; W = W1; K = K1; kt = s * BK; }
    else          { A = A2; W = W2; K = K2; kt = (s - nst1) * BK; }
    #pragma unroll
    for (int i = 0; i < 4; ++i) {                 // A tile: 128x64 = 1024 x v8bf
      int idx = tid + i * 256;
      int r = idx >> 3, g = idx & 7;
      async_copy_b128((unsigned)(size_t)&Alds[buf][r * BKP + g * 8],
                      A + (size_t)(bm + r) * K + kt + g * 8);
    }
    #pragma unroll
    for (int i = 0; i < 4; ++i) {                 // W tile: 128x64 = 1024 x v8bf
      int idx = tid + i * 256;
      int r = idx >> 3, g = idx & 7;
      async_copy_b128((unsigned)(size_t)&Wlds[buf][r * BKP + g * 8],
                      W + (size_t)(bn + r) * K + kt + g * 8);
    }
  };

  issue_stage(0, 0);
  wait_async0();
  __syncthreads();

  #pragma unroll 1
  for (int s = 0; s < nst; ++s) {
    const int buf = s & 1;
    if (s + 1 < nst) issue_stage(s + 1, buf ^ 1);   // prefetch next stage (DMA)

    const bf16_t* Ab = Alds[buf];
    const bf16_t* Wb = Wlds[buf];
    #pragma unroll
    for (int ks = 0; ks < BK; ks += 32) {
      v16bf afr[2], wfr[4];
      // A fragment 16x32: lane l15 = M, hl selects K[hl*8..+7] / K[16+hl*8..+7]
      #pragma unroll
      for (int im = 0; im < 2; ++im) {
        const bf16_t* p = &Ab[(wm * 32 + im * 16 + l15) * BKP + ks + hl * 8];
        afr[im] = frag_cat(*(const v8bf*)p, *(const v8bf*)(p + 16));
      }
      // B fragment 32x16 from W^T: lane holds W[n][ks+hl*8 .. +7] / W[n][ks+16+hl*8 .. +7]
      #pragma unroll
      for (int jn = 0; jn < 4; ++jn) {
        const bf16_t* p = &Wb[(wn * 64 + jn * 16 + l15) * BKP + ks + hl * 8];
        wfr[jn] = frag_cat(*(const v8bf*)p, *(const v8bf*)(p + 16));
      }
      #pragma unroll
      for (int im = 0; im < 2; ++im)
        #pragma unroll
        for (int jn = 0; jn < 4; ++jn)
          acc[im][jn] = __builtin_amdgcn_wmma_f32_16x16x32_bf16(
              false, afr[im], false, wfr[jn], (short)0, acc[im][jn],
              false, false);
    }

    wait_async0();        // next stage's DMA landed
    __syncthreads();      // all waves done reading `buf` before it is reused
  }

  // epilogue: C/D layout — VGPR r: M = r + 8*hl, N = lane&15
  #pragma unroll
  for (int im = 0; im < 2; ++im) {
    #pragma unroll
    for (int jn = 0; jn < 4; ++jn) {
      int n = bn + wn * 64 + jn * 16 + l15;
      float bias = (b1 ? b1[n] : 0.f) + (b2 ? b2[n] : 0.f);
      int mb = bm + wm * 32 + im * 16 + hl * 8;
      #pragma unroll
      for (int r = 0; r < 8; ++r) {
        float v = acc[im][jn][r] + bias;
        if (relu_act) v = fmaxf(v, 0.f);
        size_t o = (size_t)(mb + r) * N + n;
        if (Cf) Cf[o] = v;
        if (Cb) Cb[o] = (bf16_t)v;
      }
    }
  }
}

__device__ __forceinline__ float sigm(float x) { return 1.f / (1.f + __expf(-x)); }

// gates (B,4H) fp32 -> h1,c1 fp32 (d_out) + h1 bf16 (next GEMM operand)
__global__ void lstm_cell_kernel(const float* __restrict__ gates,
                                 const float* __restrict__ c0,
                                 float* __restrict__ h1, float* __restrict__ c1,
                                 bf16_t* __restrict__ hb, int BH, int H)
{
  int stride = gridDim.x * blockDim.x;
  for (int i = blockIdx.x * blockDim.x + threadIdx.x; i < BH; i += stride) {
    int b = i / H, n = i - b * H;
    const float* g = gates + (size_t)b * (4 * H);
    float ig = sigm(g[n]);
    float fg = sigm(g[H + n]);
    float gg = tanhf(g[2 * H + n]);
    float og = sigm(g[3 * H + n]);
    float c = fg * c0[i] + ig * gg;
    float h = og * tanhf(c);
    c1[i] = c;
    h1[i] = h;
    hb[i] = (bf16_t)h;
  }
}

__global__ void embed_kernel(const int* __restrict__ ids, const float* __restrict__ emb,
                             bf16_t* __restrict__ x, int BE, int E)
{
  int stride = gridDim.x * blockDim.x;
  for (int i = blockIdx.x * blockDim.x + threadIdx.x; i < BE; i += stride) {
    int b = i / E, e = i - b * E;
    x[i] = (bf16_t)emb[(size_t)ids[b] * E + e];
  }
}

__global__ void cvt_kernel(const float* __restrict__ src, bf16_t* __restrict__ dst, int n)
{
  int stride = gridDim.x * blockDim.x;
  for (int i = blockIdx.x * blockDim.x + threadIdx.x; i < n; i += stride)
    dst[i] = (bf16_t)src[i];
}

extern "C" void kernel_launch(void* const* d_in, const int* in_sizes, int n_in,
                              void* d_out, int out_size, void* d_ws, size_t ws_size,
                              hipStream_t stream)
{
  constexpr int B = 8192, V = 1024, E = 640, H = 640, L = 2, D = 512, J = 640;
  (void)in_sizes; (void)n_in; (void)out_size; (void)ws_size;

  const int*   ids    = (const int*)  d_in[0];
  const float* h0     = (const float*)d_in[1];
  const float* c0     = (const float*)d_in[2];
  const float* enc    = (const float*)d_in[3];
  const float* emb    = (const float*)d_in[4];
  const float* w_ih   = (const float*)d_in[5];
  const float* w_hh   = (const float*)d_in[6];
  const float* b_ih   = (const float*)d_in[7];
  const float* b_hh   = (const float*)d_in[8];
  const float* w_enc  = (const float*)d_in[9];
  const float* b_enc  = (const float*)d_in[10];
  const float* w_pred = (const float*)d_in[11];
  const float* b_pred = (const float*)d_in[12];
  const float* w_out  = (const float*)d_in[13];
  const float* b_out  = (const float*)d_in[14];

  float* logits = (float*)d_out;                  // (B,V)
  float* h1     = logits + (size_t)B * V;         // (L,B,H)
  float* c1     = h1 + (size_t)L * B * H;         // (L,B,H)

  // workspace carve-up (256B aligned)
  char* ws = (char*)d_ws;
  size_t off = 0;
  auto carve = [&](size_t bytes) -> char* {
    char* p = ws + off; off += (bytes + 255) & ~(size_t)255; return p;
  };
  bf16_t* xb     = (bf16_t*)carve((size_t)B * E * 2);       // embedded tokens
  bf16_t* h0b    = (bf16_t*)carve((size_t)L * B * H * 2);   // h0 both layers
  bf16_t* encb   = (bf16_t*)carve((size_t)B * D * 2);
  bf16_t* predb  = (bf16_t*)carve((size_t)2 * B * H * 2);   // h1 of L0, L1 (bf16)
  bf16_t* zb     = (bf16_t*)carve((size_t)B * J * 2);
  float*  gates  = (float*) carve((size_t)B * 4 * H * 4);   // reused across layers
  bf16_t* wihb   = (bf16_t*)carve((size_t)L * 4 * H * H * 2);
  bf16_t* whhb   = (bf16_t*)carve((size_t)L * 4 * H * H * 2);
  bf16_t* wencb  = (bf16_t*)carve((size_t)J * D * 2);
  bf16_t* wpredb = (bf16_t*)carve((size_t)J * H * 2);
  bf16_t* woutb  = (bf16_t*)carve((size_t)V * J * 2);

  dim3 blk(256);
  dim3 gcvt(2048);

  // fp32 -> bf16 staging
  cvt_kernel<<<gcvt, blk, 0, stream>>>(w_ih,   wihb,   L * 4 * H * H);
  cvt_kernel<<<gcvt, blk, 0, stream>>>(w_hh,   whhb,   L * 4 * H * H);
  cvt_kernel<<<gcvt, blk, 0, stream>>>(w_enc,  wencb,  J * D);
  cvt_kernel<<<gcvt, blk, 0, stream>>>(w_pred, wpredb, J * H);
  cvt_kernel<<<gcvt, blk, 0, stream>>>(w_out,  woutb,  V * J);
  cvt_kernel<<<gcvt, blk, 0, stream>>>(h0,     h0b,    L * B * H);
  cvt_kernel<<<gcvt, blk, 0, stream>>>(enc,    encb,   B * D);
  embed_kernel<<<gcvt, blk, 0, stream>>>(ids, emb, xb, B * E, E);

  const size_t BH = (size_t)B * H;
  dim3 gcell((B * H) / 256);

  // ---- LSTM layer 0: gates = x*Wih0^T + h0[0]*Whh0^T + b_ih0 + b_hh0
  gemm_dual_kernel<<<dim3((4 * H) / BN, B / BM), blk, 0, stream>>>(
      xb, wihb, E, h0b, whhb, H, b_ih, b_hh,
      gates, nullptr, B, 4 * H, 0);
  lstm_cell_kernel<<<gcell, blk, 0, stream>>>(
      gates, c0, h1, c1, predb, B * H, H);

  // ---- LSTM layer 1
  gemm_dual_kernel<<<dim3((4 * H) / BN, B / BM), blk, 0, stream>>>(
      predb, wihb + (size_t)4 * H * H, H,
      h0b + BH, whhb + (size_t)4 * H * H, H,
      b_ih + 4 * H, b_hh + 4 * H,
      gates, nullptr, B, 4 * H, 0);
  lstm_cell_kernel<<<gcell, blk, 0, stream>>>(
      gates, c0 + BH, h1 + BH, c1 + BH, predb + BH, B * H, H);

  // ---- joint: z = relu(enc*We^T + pred*Wp^T + be + bp), bf16 only
  gemm_dual_kernel<<<dim3(J / BN, B / BM), blk, 0, stream>>>(
      encb, wencb, D, predb + BH, wpredb, H, b_enc, b_pred,
      nullptr, zb, B, J, 1);

  // ---- logits = z*Wout^T + b_out (fp32 to d_out)
  gemm_dual_kernel<<<dim3(V / BN, B / BM), blk, 0, stream>>>(
      zb, woutb, J, nullptr, nullptr, 0, b_out, nullptr,
      logits, nullptr, B, V, 0);
}
